// QuantLinear_36223754174464
// MI455X (gfx1250) — matrix operand order repaired and checked by reading, not compiled, not run on hardware
//
#include <hip/hip_runtime.h>
#include <hip/hip_bf16.h>

// GPTQ W4A16 GEMM for MI455X (gfx1250, wave32, WMMA).
// D = x[16,8192] @ dequant(qweight[1024,28672] int4x8, scales[64,28672])
//
// Memory-bound: ~127MB/call -> ~5.5us floor @ 23.3TB/s; needs ~1.4 PFLOP/s
// => v_wmma_f32_16x16x32_f16 (M=16 == WMMA tile M), f32 accumulation.
// Dequant via 0x6400 packed-f16 trick + v_pk_fma_f16 (co-executes with WMMA).
//
// B operand layout (dense 32x16 f16, contiguous K order chosen by this kernel):
//   lanes 0-15 hold K=0..15 (VGPR j: K=2j,2j+1), lanes 16-31 hold K=16..31.
// A operand layout (16x32 f16, same contiguous K order as B):
//   lanes 0-15 hold K=0..15, lanes 16-31 hold K=16..31.
// A shared K-permutation sigma(8a+b) = 8a + P[b], P={0,4,1,5,2,6,3,7}, is
// applied to BOTH operands so B dequant extracts nibble pairs (j, j+4) with a
// single and_or + pk_fma per two halves. GEMM is K-order invariant => exact.

typedef __attribute__((ext_vector_type(16))) _Float16 v16h;
typedef __attribute__((ext_vector_type(2)))  _Float16 v2h;
typedef __attribute__((ext_vector_type(8)))  float    v8f;
typedef __attribute__((ext_vector_type(4)))  float    v4f;

#define MM   16
#define KK   8192
#define NN   28672
#define GRP  128
#define NT   2     // 16-col output tiles per wave
#define WVS  2     // waves per block

__global__ __launch_bounds__(WVS * 32)
void gptq_w4a16_wmma(const float* __restrict__ x,
                     const int*   __restrict__ qw,
                     const float* __restrict__ scales,
                     float* __restrict__ out)
{
    const int lane = threadIdx.x & 31;
    const int wave = threadIdx.x >> 5;
    const int col  = lane & 15;   // N column within tile / A row M
    const int kh   = lane >> 4;   // lane-half select
    const int n0   = (blockIdx.x * WVS + wave) * (NT * 16);

    const float* xrow = x + col * KK;   // A-matrix row M = lane&15

    v8f acc[NT];
#pragma unroll
    for (int t = 0; t < NT; ++t) acc[t] = (v8f)0.0f;

    constexpr int P[8] = {0, 4, 1, 5, 2, 6, 3, 7};

    // Per-lane prefetch target: together the 32 lanes cover one group's
    // 16 qweight rows x 32 cols (2KB) for this wave.
    const int pfrow = lane & 15;
    const int pfcol = n0 + (lane >> 4) * 16;

    for (int g = 0; g < KK / GRP; ++g) {
        // Prefetch next group's qweight footprint (speculative, streaming).
        {
            const int gn = (g + 1 < KK / GRP) ? (g + 1) : g;
            __builtin_prefetch(qw + (gn * (GRP / 8) + pfrow) * NN + pfcol, 0, 0);
        }

        // Per-group, per-column scale (group aligned to 4 K-chunks of 32).
        v2h sc2[NT], sb2[NT];
#pragma unroll
        for (int t = 0; t < NT; ++t) {
            float s = scales[g * NN + n0 + t * 16 + col];
            _Float16 sh = (_Float16)s;
            _Float16 bh = (_Float16)(-1032.0f * s);  // -(1024+zp)*scale
            sc2[t][0] = sh; sc2[t][1] = sh;
            sb2[t][0] = bh; sb2[t][1] = bh;
        }
#pragma unroll
        for (int c = 0; c < GRP / 32; ++c) {
            const int kb = g * GRP + c * 32;

            // ---- A fragment: two contiguous 32B loads, permute on cvt ----
            v4f a0 = *(const v4f*)(xrow + kb + 16 * kh);
            v4f a1 = *(const v4f*)(xrow + kb + 16 * kh + 4);
            v4f a2 = *(const v4f*)(xrow + kb + 16 * kh + 8);
            v4f a3 = *(const v4f*)(xrow + kb + 16 * kh + 12);
            float lo8[8] = {a0.x, a0.y, a0.z, a0.w, a1.x, a1.y, a1.z, a1.w};
            float hi8[8] = {a2.x, a2.y, a2.z, a2.w, a3.x, a3.y, a3.z, a3.w};
            v16h afrag;
#pragma unroll
            for (int s = 0; s < 8; ++s) {
                afrag[s]     = (_Float16)lo8[P[s]];
                afrag[s + 8] = (_Float16)hi8[P[s]];
            }

            // ---- B fragments: lane-half holds contiguous K 16-block =>
            //      two consecutive qweight rows per lane, coalesced dwords ----
            const int kr8 = (kb >> 3) + 2 * kh;
#pragma unroll
            for (int t = 0; t < NT; ++t) {
                const int cn = n0 + t * 16 + col;
                uint32_t w0 = (uint32_t)qw[(kr8 + 0) * NN + cn];
                uint32_t w1 = (uint32_t)qw[(kr8 + 1) * NN + cn];
                v16h bfrag;
#pragma unroll
                for (int j = 0; j < 4; ++j) {
                    union { uint32_t u; v2h h; } c0, c1;
                    c0.u = ((w0 >> (4 * j)) & 0x000F000Fu) | 0x64006400u;
                    c1.u = ((w1 >> (4 * j)) & 0x000F000Fu) | 0x64006400u;
                    v2h r0 = c0.h * sc2[t] + sb2[t];   // v_pk_fma_f16
                    v2h r1 = c1.h * sc2[t] + sb2[t];
                    bfrag[2 * j]         = r0[0];
                    bfrag[2 * j + 1]     = r0[1];
                    bfrag[8 + 2 * j]     = r1[0];
                    bfrag[8 + 2 * j + 1] = r1[1];
                }
                // D = A x B + C   (v_wmma_f32_16x16x32_f16)
                acc[t] = __builtin_amdgcn_wmma_f32_16x16x32_f16(
                    false, afrag, false, bfrag, (short)0, acc[t], false, false);
            }
        }
    }

    // ---- store C/D tile: VGPR r, lanes 0-15 -> M=r, lanes 16-31 -> M=r+8 ----
#pragma unroll
    for (int t = 0; t < NT; ++t) {
#pragma unroll
        for (int r = 0; r < 8; ++r) {
            const int m = r + 8 * kh;
            out[m * NN + n0 + t * 16 + col] = acc[t][r];
        }
    }
}

extern "C" void kernel_launch(void* const* d_in, const int* in_sizes, int n_in,
                              void* d_out, int out_size, void* d_ws, size_t ws_size,
                              hipStream_t stream) {
    (void)in_sizes; (void)n_in; (void)d_ws; (void)ws_size; (void)out_size;
    const float* x      = (const float*)d_in[0];
    const int*   qw     = (const int*)d_in[1];
    const float* scales = (const float*)d_in[2];
    float*       out    = (float*)d_out;

    const int blocks = NN / (NT * 16 * WVS);   // 448 blocks x 64 threads = 896 waves
    gptq_w4a16_wmma<<<blocks, WVS * 32, 0, stream>>>(x, qw, scales, out);
}
